// EMNNGNN_84387517432503
// MI455X (gfx1250) — matrix operand order repaired and checked by reading, not compiled, not run on hardware
//
#include <hip/hip_runtime.h>
#include <hip/hip_bf16.h>

// ---------------------------------------------------------------------------
// EMNN GNN on MI455X (gfx1250, wave32, WMMA).
// Everything (29 MB of state) is L2-resident -> recompute-over-spill strategy.
// One wave per 16-edge tile; all matmuls via v_wmma_f32_16x16x32_bf16.
// ---------------------------------------------------------------------------

typedef __attribute__((ext_vector_type(16))) __bf16 v16bf;
typedef __attribute__((ext_vector_type(8)))  float  v8f;

#define NNODES 3000
#define NEDGES 16000
#define FDIM   32
#define NSTEPS 6
#define NTILES (NEDGES / 16)

__device__ __forceinline__ v8f wmma_bf16(v16bf a, v16bf b, v8f c) {
  // D = A(16x32) * B(32x16) + C, f32 accumulate
  return __builtin_amdgcn_wmma_f32_16x16x32_bf16(false, a, false, b,
                                                 (short)0, c, false, false);
}

// Build a bf16 A-fragment (16x32, row-major f32 in LDS) per ISA 7.12.2 layout:
// lane m=lane&15, half=lane>>4; VGPR v holds K pair kb,kb+1.
__device__ __forceinline__ v16bf make_A_lds(const float* m, int lane) {
  const int half = lane >> 4, r = lane & 15;
  v16bf a;
#pragma unroll
  for (int v = 0; v < 8; ++v) {
    const int kb = ((v >> 2) << 4) + ((v & 3) << 1) + (half << 3);
    a[2 * v]     = (__bf16)m[r * FDIM + kb];
    a[2 * v + 1] = (__bf16)m[r * FDIM + kb + 1];
  }
  return a;
}

// Pre-packed B fragment: 512 bf16 per 32x16 chunk, laid out so each lane does
// one coalesced 32B load (lowers to global_load_b128 x2).
__device__ __forceinline__ v16bf load_frag(const __bf16* p, int chunk, int lane) {
  return ((const v16bf*)(p + (size_t)chunk * 512))[lane];
}

// ---------------------------------------------------------------------------
__global__ void zero_f32_kernel(float* __restrict__ p, int n) {
  int i = blockIdx.x * blockDim.x + threadIdx.x;
  if (i < n) p[i] = 0.f;
}

// Repack an f32 weight matrix into bf16 WMMA-B fragment layout.
// transposed==0: B(k,n) = w[k*ld + c*16 + n]   (w is K x N, row-major)
// transposed==1: B(k,n) = w[(c*16+n)*ld + k]   (w is N x K, we need w^T)
__global__ void pack_b_kernel(const float* __restrict__ w, __bf16* __restrict__ out,
                              int ld, int nchunks, int transposed) {
  int idx = blockIdx.x * blockDim.x + threadIdx.x;
  if (idx >= nchunks * 512) return;
  const int c = idx >> 9;
  const int l = (idx >> 4) & 31;
  const int i = idx & 15;
  const int k = ((i >> 3) << 4) + ((l >> 4) << 3) + (i & 7);
  const int n = l & 15;
  const float v = transposed ? w[(c * 16 + n) * ld + k]
                             : w[k * ld + c * 16 + n];
  out[idx] = (__bf16)v;
}

// ief = relu(concat(nf[src], nf[dst], ef) @ Wi); hstate = ief   (one-time)
__global__ void init_ief_kernel(const float* __restrict__ node_feats,
                                const float* __restrict__ edge_feats,
                                const int* __restrict__ src,
                                const int* __restrict__ dst,
                                const float* __restrict__ Wi,
                                float* __restrict__ ief,
                                float* __restrict__ hstate) {
  int idx = blockIdx.x * blockDim.x + threadIdx.x;
  if (idx >= NEDGES * FDIM) return;
  const int e = idx >> 5, f = idx & 31;
  const float* ns = node_feats + src[e] * 64;
  const float* nd = node_feats + dst[e] * 64;
  const float* ef = edge_feats + e * 16;
  float acc = 0.f;
#pragma unroll 4
  for (int k = 0; k < 64; ++k) acc += ns[k] * Wi[k * FDIM + f];
#pragma unroll 4
  for (int k = 0; k < 64; ++k) acc += nd[k] * Wi[(64 + k) * FDIM + f];
#pragma unroll
  for (int k = 0; k < 16; ++k) acc += ef[k] * Wi[(128 + k) * FDIM + f];
  acc = fmaxf(acc, 0.f);
  ief[idx] = acc;
  hstate[idx] = acc;
}

// ---------------------------------------------------------------------------
// Pass A: per 16-edge tile, compute exp(wa*h) and exp(wa*h)*wm*h per chunk and
// segment-sum them into node accumulators with f32 atomics (L2-resident).
__global__ __launch_bounds__(32) void passA_kernel(
    const float* __restrict__ hstate, const int* __restrict__ dst,
    const __bf16* __restrict__ pw1m, const float* __restrict__ m_b1,
    const __bf16* __restrict__ pw2m, const float* __restrict__ m_b2,
    const __bf16* __restrict__ pw1a, const float* __restrict__ a_b1,
    const __bf16* __restrict__ pw2a, const float* __restrict__ a_b2,
    float* __restrict__ sumexp_node, float* __restrict__ mnode) {
  __shared__ float h_lds[16 * FDIM];
  __shared__ float t1_lds[16 * FDIM];
  __shared__ float t1a_lds[16 * FDIM];
  __shared__ int   dst_lds[16];
  const int lane = threadIdx.x;
  const int e0 = blockIdx.x * 16;
  for (int t = lane; t < 16 * FDIM; t += 32) h_lds[t] = hstate[e0 * FDIM + t];
  if (lane < 16) dst_lds[lane] = dst[e0 + lane];
  __syncthreads();

  const int n_ = lane & 15, hl = lane >> 4;
  v16bf hA = make_A_lds(h_lds, lane);
#pragma unroll
  for (int hn = 0; hn < 2; ++hn) {  // layer-1 of both MLPs (relu)
    v8f d0 = {}; d0 = wmma_bf16(hA, load_frag(pw1m, hn, lane), d0);
    v8f d1 = {}; d1 = wmma_bf16(hA, load_frag(pw1a, hn, lane), d1);
    const float b0 = m_b1[hn * 16 + n_];
    const float b1 = a_b1[hn * 16 + n_];
#pragma unroll
    for (int v = 0; v < 8; ++v) {
      t1_lds [(v + 8 * hl) * FDIM + hn * 16 + n_] = fmaxf(d0[v] + b0, 0.f);
      t1a_lds[(v + 8 * hl) * FDIM + hn * 16 + n_] = fmaxf(d1[v] + b1, 0.f);
    }
  }
  __syncthreads();
  v16bf t1A  = make_A_lds(t1_lds, lane);
  v16bf t1aA = make_A_lds(t1a_lds, lane);

  int dsts[8];
#pragma unroll
  for (int v = 0; v < 8; ++v) dsts[v] = dst_lds[v + 8 * hl];

  for (int c = 0; c < 64; ++c) {          // 64 chunks of the 32x1024 layer
    const int col0 = c * 16, i = c >> 1;  // flat col = i*32 + j
    v8f wm = {}; wm = wmma_bf16(t1A,  load_frag(pw2m, c, lane), wm);
    v8f wa = {}; wa = wmma_bf16(t1aA, load_frag(pw2a, c, lane), wa);
    const float bm = m_b2[col0 + n_];
    const float ba = a_b2[col0 + n_];
#pragma unroll
    for (int v = 0; v < 8; ++v) {
      const int row = v + 8 * hl;
      const float hv  = h_lds[row * FDIM + i];
      const float wmv = wm[v] + bm;
      const float wav = wa[v] + ba;
      const float ee2 = __expf(wav * hv);
      const float h1  = ee2 * (wmv * hv);
      const int base = dsts[v] * 1024 + col0 + n_;
      atomicAdd(&sumexp_node[base], ee2);
      atomicAdd(&mnode[base], h1);
    }
  }
}

// ---------------------------------------------------------------------------
// Pass B: recompute chunks (cheaper than spilling 128MB/step), gather node
// sums by src, finish message h2, reduce over i, fused GRU, write new state.
__global__ __launch_bounds__(32) void passB_kernel(
    float* __restrict__ hstate, const float* __restrict__ ief,
    const int* __restrict__ src,
    const __bf16* __restrict__ pw1m, const float* __restrict__ m_b1,
    const __bf16* __restrict__ pw2m, const float* __restrict__ m_b2,
    const __bf16* __restrict__ pw1a, const float* __restrict__ a_b1,
    const __bf16* __restrict__ pw2a, const float* __restrict__ a_b2,
    const __bf16* __restrict__ pwih, const float* __restrict__ bih,
    const __bf16* __restrict__ pwhh, const float* __restrict__ bhh,
    const float* __restrict__ sumexp_node, const float* __restrict__ mnode) {
  __shared__ float h_lds[16 * FDIM];
  __shared__ float i_lds[16 * FDIM];
  __shared__ float t1_lds[16 * FDIM];
  __shared__ float t1a_lds[16 * FDIM];
  __shared__ float x_lds[16 * FDIM];
  __shared__ int   src_lds[16];
  const int lane = threadIdx.x;
  const int e0 = blockIdx.x * 16;
  for (int t = lane; t < 16 * FDIM; t += 32) {
    h_lds[t] = hstate[e0 * FDIM + t];
    i_lds[t] = ief[e0 * FDIM + t];
  }
  if (lane < 16) src_lds[lane] = src[e0 + lane];
  __syncthreads();

  const int n_ = lane & 15, hl = lane >> 4;
  v16bf hA = make_A_lds(h_lds, lane);
#pragma unroll
  for (int hn = 0; hn < 2; ++hn) {
    v8f d0 = {}; d0 = wmma_bf16(hA, load_frag(pw1m, hn, lane), d0);
    v8f d1 = {}; d1 = wmma_bf16(hA, load_frag(pw1a, hn, lane), d1);
    const float b0 = m_b1[hn * 16 + n_];
    const float b1 = a_b1[hn * 16 + n_];
#pragma unroll
    for (int v = 0; v < 8; ++v) {
      t1_lds [(v + 8 * hl) * FDIM + hn * 16 + n_] = fmaxf(d0[v] + b0, 0.f);
      t1a_lds[(v + 8 * hl) * FDIM + hn * 16 + n_] = fmaxf(d1[v] + b1, 0.f);
    }
  }
  __syncthreads();
  v16bf t1A  = make_A_lds(t1_lds, lane);
  v16bf t1aA = make_A_lds(t1a_lds, lane);

  int srcs[8];
#pragma unroll
  for (int v = 0; v < 8; ++v) srcs[v] = src_lds[v + 8 * hl];

  float rst0[8] = {0,0,0,0,0,0,0,0};   // j in [0,16)  accum over i
  float rst1[8] = {0,0,0,0,0,0,0,0};   // j in [16,32) accum over i
  for (int c = 0; c < 64; ++c) {
    const int col0 = c * 16, i = c >> 1;
    v8f wm = {}; wm = wmma_bf16(t1A,  load_frag(pw2m, c, lane), wm);
    v8f wa = {}; wa = wmma_bf16(t1aA, load_frag(pw2a, c, lane), wa);
    const float bm = m_b2[col0 + n_];
    const float ba = a_b2[col0 + n_];
#pragma unroll
    for (int v = 0; v < 8; ++v) {
      const int row = v + 8 * hl;
      const float hv  = h_lds[row * FDIM + i];
      const float iv  = i_lds[row * FDIM + i];
      const float wmv = wm[v] + bm;
      const float wav = wa[v] + ba;
      const float ee2 = __expf(wav * hv);
      const float iex = __expf(wav * iv);
      const float h1  = ee2 * (wmv * hv);
      const float ih1 = iex * (wmv * iv);
      const int base = srcs[v] * 1024 + col0 + n_;
      const float ss = sumexp_node[base];
      const float ms = mnode[base];
      const float h2 = (ms - h1 + ih1) / (ss - ee2 + iex);
      if (c & 1) rst1[v] += h2; else rst0[v] += h2;
    }
  }
#pragma unroll
  for (int v = 0; v < 8; ++v) {
    const int row = v + 8 * hl;
    x_lds[row * FDIM + n_]      = fmaxf(rst0[v], 0.f);
    x_lds[row * FDIM + 16 + n_] = fmaxf(rst1[v], 0.f);
  }
  __syncthreads();
  v16bf xA = make_A_lds(x_lds, lane);   // GRU input; hA is still hidden

#pragma unroll
  for (int cc = 0; cc < 2; ++cc) {      // gate columns cc*16 .. cc*16+15
    v8f gr = {}; gr = wmma_bf16(xA, load_frag(pwih, cc,     lane), gr);
    v8f hr = {}; hr = wmma_bf16(hA, load_frag(pwhh, cc,     lane), hr);
    v8f gz = {}; gz = wmma_bf16(xA, load_frag(pwih, 2 + cc, lane), gz);
    v8f hz = {}; hz = wmma_bf16(hA, load_frag(pwhh, 2 + cc, lane), hz);
    v8f gn = {}; gn = wmma_bf16(xA, load_frag(pwih, 4 + cc, lane), gn);
    v8f hv = {}; hv = wmma_bf16(hA, load_frag(pwhh, 4 + cc, lane), hv);
    const float br_i = bih[cc * 16 + n_],      br_h = bhh[cc * 16 + n_];
    const float bz_i = bih[32 + cc * 16 + n_], bz_h = bhh[32 + cc * 16 + n_];
    const float bn_i = bih[64 + cc * 16 + n_], bn_h = bhh[64 + cc * 16 + n_];
#pragma unroll
    for (int v = 0; v < 8; ++v) {
      const int row = v + 8 * hl;
      const float rpre = (gr[v] + br_i) + (hr[v] + br_h);
      const float zpre = (gz[v] + bz_i) + (hz[v] + bz_h);
      const float r = 1.f / (1.f + __expf(-rpre));
      const float z = 1.f / (1.f + __expf(-zpre));
      const float npre = (gn[v] + bn_i) + r * (hv[v] + bn_h);
      const float t = __expf(-2.f * npre);
      const float nn = (1.f - t) / (1.f + t);
      const float hold = h_lds[row * FDIM + cc * 16 + n_];
      hstate[(e0 + row) * FDIM + cc * 16 + n_] = (1.f - z) * nn + z * hold;
    }
  }
}

// out[dst[e], f] += h[e, f]
__global__ void scatter_out_kernel(const float* __restrict__ h,
                                   const int* __restrict__ dst,
                                   float* __restrict__ out) {
  int idx = blockIdx.x * blockDim.x + threadIdx.x;
  if (idx >= NEDGES * FDIM) return;
  const int e = idx >> 5, f = idx & 31;
  atomicAdd(&out[dst[e] * FDIM + f], h[idx]);
}

// ---------------------------------------------------------------------------
extern "C" void kernel_launch(void* const* d_in, const int* in_sizes, int n_in,
                              void* d_out, int out_size, void* d_ws, size_t ws_size,
                              hipStream_t stream) {
  const float* node_feats = (const float*)d_in[0];
  const float* edge_feats = (const float*)d_in[1];
  const int*   src        = (const int*)d_in[2];
  const int*   dst        = (const int*)d_in[3];
  const float* Wi         = (const float*)d_in[4];
  const float* m_w1       = (const float*)d_in[5];
  const float* m_b1       = (const float*)d_in[6];
  const float* m_w2       = (const float*)d_in[7];
  const float* m_b2       = (const float*)d_in[8];
  const float* a_w1       = (const float*)d_in[9];
  const float* a_b1       = (const float*)d_in[10];
  const float* a_w2       = (const float*)d_in[11];
  const float* a_b2       = (const float*)d_in[12];
  const float* gru_wih    = (const float*)d_in[13];
  const float* gru_whh    = (const float*)d_in[14];
  const float* gru_bih    = (const float*)d_in[15];
  const float* gru_bhh    = (const float*)d_in[16];

  char* ws = (char*)d_ws;
  size_t off = 0;
  float*  ief     = (float*)(ws + off); off += (size_t)NEDGES * FDIM * 4;   // 2.048 MB
  float*  hstate  = (float*)(ws + off); off += (size_t)NEDGES * FDIM * 4;   // 2.048 MB
  float*  sumexp  = (float*)(ws + off); off += (size_t)NNODES * 1024 * 4;   // 12.288 MB
  float*  mnode   = (float*)(ws + off); off += (size_t)NNODES * 1024 * 4;   // 12.288 MB
  __bf16* pw1m    = (__bf16*)(ws + off); off += 2  * 512 * 2;
  __bf16* pw1a    = (__bf16*)(ws + off); off += 2  * 512 * 2;
  __bf16* pw2m    = (__bf16*)(ws + off); off += 64 * 512 * 2;
  __bf16* pw2a    = (__bf16*)(ws + off); off += 64 * 512 * 2;
  __bf16* pwih    = (__bf16*)(ws + off); off += 6  * 512 * 2;
  __bf16* pwhh    = (__bf16*)(ws + off); off += 6  * 512 * 2;

  // One-time weight repack into bf16 WMMA-B fragment layout.
  pack_b_kernel<<<(2  * 512 + 255) / 256, 256, 0, stream>>>(m_w1,    pw1m, 32,   2,  0);
  pack_b_kernel<<<(2  * 512 + 255) / 256, 256, 0, stream>>>(a_w1,    pw1a, 32,   2,  0);
  pack_b_kernel<<<(64 * 512 + 255) / 256, 256, 0, stream>>>(m_w2,    pw2m, 1024, 64, 0);
  pack_b_kernel<<<(64 * 512 + 255) / 256, 256, 0, stream>>>(a_w2,    pw2a, 1024, 64, 0);
  pack_b_kernel<<<(6  * 512 + 255) / 256, 256, 0, stream>>>(gru_wih, pwih, 32,   6,  1);
  pack_b_kernel<<<(6  * 512 + 255) / 256, 256, 0, stream>>>(gru_whh, pwhh, 32,   6,  1);

  // ief = relu(h0 @ Wi); hstate = ief
  init_ief_kernel<<<(NEDGES * FDIM + 255) / 256, 256, 0, stream>>>(
      node_feats, edge_feats, src, dst, Wi, ief, hstate);

  const int nacc = NNODES * 1024 * 2;  // sumexp + mnode are contiguous
  for (int step = 0; step < NSTEPS; ++step) {
    zero_f32_kernel<<<(nacc + 255) / 256, 256, 0, stream>>>(sumexp, nacc);
    passA_kernel<<<NTILES, 32, 0, stream>>>(
        hstate, dst, pw1m, m_b1, pw2m, m_b2, pw1a, a_b1, pw2a, a_b2,
        sumexp, mnode);
    passB_kernel<<<NTILES, 32, 0, stream>>>(
        hstate, ief, src, pw1m, m_b1, pw2m, m_b2, pw1a, a_b1, pw2a, a_b2,
        pwih, gru_bih, pwhh, gru_bhh, sumexp, mnode);
  }

  zero_f32_kernel<<<(out_size + 255) / 256, 256, 0, stream>>>((float*)d_out, out_size);
  scatter_out_kernel<<<(NEDGES * FDIM + 255) / 256, 256, 0, stream>>>(
      hstate, dst, (float*)d_out);
}
